// MultiHeadSelfAttention_67413806678621
// MI455X (gfx1250) — matrix operand order repaired and checked by reading, not compiled, not run on hardware
//
#include <hip/hip_runtime.h>

typedef __attribute__((ext_vector_type(16))) __bf16    v16bf;
typedef __attribute__((ext_vector_type(8)))  float     v8f;
typedef __attribute__((ext_vector_type(4)))  unsigned  v4u;
typedef __attribute__((ext_vector_type(8)))  unsigned  v8u;

// Problem constants (from reference): N=8, T=2048, D=512, H=8, DK=64
constexpr int Nb  = 8;
constexpr int Ts  = 2048;
constexpr int Dm  = 512;
constexpr int Hh  = 8;
constexpr int DKk = 64;
constexpr int NT  = Nb * Ts;   // 16384 rows

// Staged B-tile LDS row stride: 64 bf16 (32 DW) + 4-DW TDM pad = 36 DW = 72 ushorts.
// 144 B rows are 16B-aligned -> ds_load_b128; starts (36*L)%64 for a half-wave
// tile all 64 banks exactly once -> conflict-free.
constexpr int BROW = 72;

// ---------------------------------------------------------------------------
// Helpers
// ---------------------------------------------------------------------------
union Frag {
    v16bf    h;    // bf16 A/B fragment (16 bf16 = 32B = 8 VGPRs)
    v8f      f;    // f32 C/D fragment  (8 f32  = 32B = 8 VGPRs)
    unsigned u[8];
    float    s[8];
};

__device__ __forceinline__ unsigned short f2bf(float x) {
    unsigned u = __float_as_uint(x);
    u += 0x7FFFu + ((u >> 16) & 1u);   // round-to-nearest-even
    return (unsigned short)(u >> 16);
}

// A fragment (16x32 bf16, ISA 7.12.2): lane holds row M = lane%16.
// 'kbase' must already be k0 + 8*(lane>>4).
__device__ __forceinline__ void load_a(Frag& a, const unsigned short* A,
                                       int lda, int mrow, int kbase) {
    const unsigned short* row = A + (size_t)mrow * lda + kbase;
#pragma unroll
    for (int j = 0; j < 8; ++j) {
        int off = 2 * j + ((j >= 4) ? 8 : 0);
        a.u[j] = *reinterpret_cast<const unsigned*>(row + off);
    }
}

// B fragment (32x16 bf16): lane L holds column N = n0+L%16, regs hold K in table order.
__device__ __forceinline__ void load_b(Frag& b, const unsigned short* B,
                                       int ldb, int k0, int n0, int lane) {
    const unsigned short* col = B + (size_t)(k0 + 8 * (lane >> 4)) * ldb + n0 + (lane & 15);
#pragma unroll
    for (int j = 0; j < 8; ++j) {
        int kk = 2 * j + ((j >= 4) ? 8 : 0);
        b.u[j] = (unsigned)col[(size_t)kk * ldb] | ((unsigned)col[(size_t)(kk + 1) * ldb] << 16);
    }
}

// B fragment from a TDM-staged LDS tile: lane holds column n0+L%16, K in table order.
__device__ __forceinline__ void lds_b(Frag& b, const unsigned short* tile,
                                      int n0, int lane) {
    const unsigned short* col = tile + 8 * (lane >> 4) * BROW + n0 + (lane & 15);
#pragma unroll
    for (int j = 0; j < 8; ++j) {
        int kk = 2 * j + ((j >= 4) ? 8 : 0);
        b.u[j] = (unsigned)col[kk * BROW] | ((unsigned)col[(kk + 1) * BROW] << 16);
    }
}

__device__ __forceinline__ v8f wmma_bf16(const Frag& a, const Frag& b, v8f c) {
    return __builtin_amdgcn_wmma_f32_16x16x32_bf16(
        false, a.h, false, b.h, (short)0, c, false, false);
}

__device__ __forceinline__ void wait_tensorcnt0() {
#if defined(__has_builtin) && __has_builtin(__builtin_amdgcn_s_wait_tensorcnt)
    __builtin_amdgcn_s_wait_tensorcnt(0);
#else
    asm volatile("s_wait_tensorcnt 0x0" ::: "memory");
#endif
}

// TDM: DMA a 2-D tile (tile_d0 x tile_d1 elements of 2 bytes) from a
// row-major [tensor_d1][stride_d0] bf16 tensor into LDS at lds_off bytes.
// D# layout per CDNA5 ISA 8.3/8.4; 2-group form (tensors up to 2D).
// Pad: +4 DWORDs after every 32 DWORDs stored -> LDS row stride 36 DWORDs.
__device__ __forceinline__ void tdm_load_2d(unsigned lds_off,
                                            const unsigned short* gptr,
                                            unsigned tensor_d0, unsigned tensor_d1,
                                            unsigned tile_d0, unsigned tile_d1,
                                            unsigned stride_d0) {
    unsigned long long ga = (unsigned long long)(uintptr_t)gptr;
    v4u g0;
    g0[0] = 1u;                                   // count=1 (valid user D#)
    g0[1] = lds_off;                              // lds_addr (bytes)
    g0[2] = (unsigned)(ga & 0xFFFFFFFFu);         // global_addr[31:0]
    g0[3] = (unsigned)((ga >> 32) & 0x01FFFFFFu)  // global_addr[56:32]
          | (2u << 30);                           // type=2 ("image")
    v8u g1;
    g1[0] = (1u << 16)                            // data_size: 2 bytes
          | (1u << 20)                            // pad_enable
          | (4u << 22)                            // pad_interval: 32 DWORDs
          | (3u << 25);                           // pad_amount: 4 DWORDs
    g1[1] = (tensor_d0 & 0xFFFFu) << 16;                                  // dim0[15:0]
    g1[2] = ((tensor_d0 >> 16) & 0xFFFFu) | ((tensor_d1 & 0xFFFFu) << 16);// dim0 hi|dim1 lo
    g1[3] = ((tensor_d1 >> 16) & 0xFFFFu) | ((tile_d0 & 0xFFFFu) << 16);  // dim1 hi|tile0
    g1[4] = tile_d1 & 0xFFFFu;                    // tile_dim1, tile_dim2=0
    g1[5] = stride_d0;                            // tensor_dim0_stride[31:0]
    g1[6] = 0u;
    g1[7] = 0u;
    asm volatile("tensor_load_to_lds %0, %1" :: "s"(g0), "s"(g1) : "memory");
}

// ---------------------------------------------------------------------------
// 0) fp32 -> bf16 conversion of x
// ---------------------------------------------------------------------------
__global__ void cvt_x_kernel(const float* __restrict__ x,
                             unsigned short* __restrict__ xb, int n) {
    int i = blockIdx.x * blockDim.x + threadIdx.x;
    if (i < n) xb[i] = f2bf(x[i]);
}

// 1) weights fp32 [n][k] -> bf16 transposed [k][n]  (B = W^T for y = x@W^T)
__global__ void cvt_w_kernel(const float* __restrict__ wq, const float* __restrict__ wk,
                             const float* __restrict__ wv, const float* __restrict__ wo,
                             unsigned short* __restrict__ wqT, unsigned short* __restrict__ wkT,
                             unsigned short* __restrict__ wvT, unsigned short* __restrict__ woT) {
    int i = blockIdx.x * blockDim.x + threadIdx.x;
    if (i >= 4 * Dm * Dm) return;
    int sel = i / (Dm * Dm);
    int rem = i % (Dm * Dm);
    int r = rem / Dm, c = rem % Dm;           // src[r][c] = W[n=r][k=c]
    const float* src = (sel == 0) ? wq : (sel == 1) ? wk : (sel == 2) ? wv : wo;
    unsigned short* dst = (sel == 0) ? wqT : (sel == 1) ? wkT : (sel == 2) ? wvT : woT;
    dst[(size_t)c * Dm + r] = f2bf(src[rem]);  // wT[k][n]
}

// ---------------------------------------------------------------------------
// 2) Fused QKV projection. Grid (Dm/64, NT/64), 256 thr = 8 waves (4M x 2N).
//    The three 32x64 B weight panels per k-step are staged into LDS by the
//    Tensor Data Mover, DOUBLE-BUFFERED: wave 0 issues the DMAs for k+32
//    right after the iteration barrier and only drains TENSORcnt at the end
//    of the iteration, so the DMA overlaps the whole compute phase.
// ---------------------------------------------------------------------------
__global__ __launch_bounds__(256) void qkv_kernel(
    const unsigned short* __restrict__ xb,
    const unsigned short* __restrict__ wqT, const unsigned short* __restrict__ wkT,
    const unsigned short* __restrict__ wvT,
    const float* __restrict__ bq, const float* __restrict__ bk, const float* __restrict__ bv,
    unsigned short* __restrict__ qh,   // [N][H][T][DK]
    unsigned short* __restrict__ kTr,  // [N][H][DK][T]
    unsigned short* __restrict__ vh) { // [N][H][T][DK]
    __shared__ unsigned short wlds[2][3][32 * BROW];  // double-buffered B panels
    const int lane = threadIdx.x & 31, wave = threadIdx.x >> 5;
    const int half = lane >> 4;
    const int m0 = blockIdx.y * 64 + (wave >> 1) * 16;
    const int nWG = blockIdx.x * 64;
    const int n0 = nWG + (wave & 1) * 32;
    const int nloc = (wave & 1) * 32;              // column offset inside LDS tile
    const int mrow = m0 + (lane & 15);

    Frag aq[2], ak[2], av[2];
#pragma unroll
    for (int s = 0; s < 2; ++s) {
#pragma unroll
        for (int r = 0; r < 8; ++r) { aq[s].s[r] = 0.f; ak[s].s[r] = 0.f; av[s].s[r] = 0.f; }
    }

    unsigned ldsa[2][3];
#pragma unroll
    for (int bsel = 0; bsel < 2; ++bsel)
#pragma unroll
        for (int w = 0; w < 3; ++w)
            ldsa[bsel][w] = (unsigned)(uintptr_t)&wlds[bsel][w][0];

    // Prologue: stage k0=0 panels into buffer 0.
    if (wave == 0) {
        tdm_load_2d(ldsa[0][0], wqT + nWG, Dm, Dm, 64, 32, Dm);
        tdm_load_2d(ldsa[0][1], wkT + nWG, Dm, Dm, 64, 32, Dm);
        tdm_load_2d(ldsa[0][2], wvT + nWG, Dm, Dm, 64, 32, Dm);
        wait_tensorcnt0();
    }
    __syncthreads();

    int cur = 0;
    for (int k0 = 0; k0 < Dm; k0 += 32) {
        const bool more = (k0 + 32) < Dm;
        // Issue next k-step's DMAs into the other buffer. Its previous readers
        // completed their ds_loads before signaling the barrier we just passed.
        if (wave == 0 && more) {
            size_t goff = (size_t)(k0 + 32) * Dm + nWG;
            tdm_load_2d(ldsa[cur ^ 1][0], wqT + goff, Dm, Dm, 64, 32, Dm);
            tdm_load_2d(ldsa[cur ^ 1][1], wkT + goff, Dm, Dm, 64, 32, Dm);
            tdm_load_2d(ldsa[cur ^ 1][2], wvT + goff, Dm, Dm, 64, 32, Dm);
        }

        // per-wave A fragment (global, unique per wave)
        Frag a; load_a(a, xb, Dm, mrow, k0 + 8 * half);
        if (more)
            __builtin_prefetch(xb + (size_t)mrow * Dm + k0 + 32, 0, 1);

        Frag b[6];
        lds_b(b[0], &wlds[cur][0][0], nloc,      lane);
        lds_b(b[1], &wlds[cur][0][0], nloc + 16, lane);
        lds_b(b[2], &wlds[cur][1][0], nloc,      lane);
        lds_b(b[3], &wlds[cur][1][0], nloc + 16, lane);
        lds_b(b[4], &wlds[cur][2][0], nloc,      lane);
        lds_b(b[5], &wlds[cur][2][0], nloc + 16, lane);

        aq[0].f = wmma_bf16(a, b[0], aq[0].f);
        aq[1].f = wmma_bf16(a, b[1], aq[1].f);
        ak[0].f = wmma_bf16(a, b[2], ak[0].f);
        ak[1].f = wmma_bf16(a, b[3], ak[1].f);
        av[0].f = wmma_bf16(a, b[4], av[0].f);
        av[1].f = wmma_bf16(a, b[5], av[1].f);

        // Drain the k+32 DMA only now: it had the whole iteration to land.
        if (wave == 0 && more) wait_tensorcnt0();
        __syncthreads();
        cur ^= 1;
    }

#pragma unroll
    for (int s = 0; s < 2; ++s) {
#pragma unroll
        for (int r = 0; r < 8; ++r) {
            int gm = m0 + r + 8 * half;          // token row in [0,16384)
            int gc = n0 + 16 * s + (lane & 15);  // feature col in [0,512)
            int nI = gm >> 11, t = gm & (Ts - 1);
            int h = gc >> 6, dk = gc & 63;
            size_t hb = ((size_t)nI * Hh + h);
            qh [(hb * Ts + t) * DKk + dk] = f2bf(aq[s].s[r] + bq[gc]);
            kTr[(hb * DKk + dk) * Ts + t] = f2bf(ak[s].s[r] + bk[gc]);
            vh [(hb * Ts + t) * DKk + dk] = f2bf(av[s].s[r] + bv[gc]);
        }
    }
}

// ---------------------------------------------------------------------------
// 3) Flash attention. Grid (T/64, N*H), 128 thr = 4 waves; each wave owns
//    16 query rows x DK=64. KV streamed in blocks of 32; scores never hit HBM.
//    V fragments are loaded BEFORE the softmax so its VALU hides their latency.
// ---------------------------------------------------------------------------
__global__ __launch_bounds__(128) void flash_kernel(
    const unsigned short* __restrict__ qh,
    const unsigned short* __restrict__ kTr,
    const unsigned short* __restrict__ vh,
    unsigned short* __restrict__ aob) {  // [N][T][D] bf16, col = h*64+dk
    __shared__ unsigned short pbuf[4 * 16 * 32];  // per-wave 16x32 bf16 P tile
    const int lane = threadIdx.x & 31, wave = threadIdx.x >> 5;
    const int half = lane >> 4;
    const int nh = blockIdx.y;             // n*H + h
    const int n = nh >> 3, h = nh & 7;
    const int qt0 = blockIdx.x * 64 + wave * 16;

    const unsigned short* Qb = qh  + (size_t)nh * Ts * DKk;
    const unsigned short* Kb = kTr + (size_t)nh * DKk * Ts;
    const unsigned short* Vb = vh  + (size_t)nh * Ts * DKk;
    unsigned short* myp = pbuf + wave * (16 * 32);

    Frag aqf[2];
    const int mrow = qt0 + (lane & 15);
    load_a(aqf[0], Qb, DKk, mrow, 0  + 8 * half);
    load_a(aqf[1], Qb, DKk, mrow, 32 + 8 * half);

    Frag o[4];
    float m_i[8], l_i[8];
#pragma unroll
    for (int r = 0; r < 8; ++r) { m_i[r] = -1e30f; l_i[r] = 0.f; }
#pragma unroll
    for (int c = 0; c < 4; ++c)
#pragma unroll
        for (int r = 0; r < 8; ++r) o[c].s[r] = 0.f;

    for (int kb = 0; kb < Ts; kb += 32) {
        // ---- S = (Q K^T): issue all 4 K-fragment loads first
        Frag bk0[2], bk1[2];
#pragma unroll
        for (int kc = 0; kc < 2; ++kc) {
            const unsigned short* krow = Kb + (size_t)(kc * 32 + 8 * half) * Ts + kb + (lane & 15);
#pragma unroll
            for (int j = 0; j < 8; ++j) {
                int kk = 2 * j + ((j >= 4) ? 8 : 0);
                bk0[kc].u[j] = (unsigned)krow[(size_t)kk * Ts] | ((unsigned)krow[(size_t)(kk + 1) * Ts] << 16);
                bk1[kc].u[j] = (unsigned)krow[(size_t)kk * Ts + 16] | ((unsigned)krow[(size_t)(kk + 1) * Ts + 16] << 16);
            }
        }
        Frag s0, s1;
#pragma unroll
        for (int r = 0; r < 8; ++r) { s0.s[r] = 0.f; s1.s[r] = 0.f; }
        s0.f = wmma_bf16(aqf[0], bk0[0], s0.f);
        s1.f = wmma_bf16(aqf[0], bk1[0], s1.f);
        s0.f = wmma_bf16(aqf[1], bk0[1], s0.f);
        s1.f = wmma_bf16(aqf[1], bk1[1], s1.f);

        // ---- issue V fragment loads NOW; softmax VALU hides their latency
        Frag bv[4];
#pragma unroll
        for (int c = 0; c < 4; ++c) {
            const unsigned short* vrow = Vb + (size_t)(kb + 8 * half) * DKk + 16 * c + (lane & 15);
#pragma unroll
            for (int j = 0; j < 8; ++j) {
                int kk = 2 * j + ((j >= 4) ? 8 : 0);
                bv[c].u[j] = (unsigned)vrow[kk * DKk] | ((unsigned)vrow[(kk + 1) * DKk] << 16);
            }
        }

        // ---- online softmax in C layout (reg r -> row r+8*half, lane%16 -> col)
#pragma unroll
        for (int r = 0; r < 8; ++r) {
            float f0 = s0.s[r] * 0.125f, f1 = s1.s[r] * 0.125f;
            float mx = fmaxf(f0, f1);
#pragma unroll
            for (int d = 1; d < 16; d <<= 1) mx = fmaxf(mx, __shfl_xor(mx, d, 32));
            float nm = fmaxf(m_i[r], mx);
            float alpha = __expf(m_i[r] - nm);
            float p0 = __expf(f0 - nm), p1 = __expf(f1 - nm);
            float rs = p0 + p1;
#pragma unroll
            for (int d = 1; d < 16; d <<= 1) rs += __shfl_xor(rs, d, 32);
            l_i[r] = l_i[r] * alpha + rs;
            m_i[r] = nm;
#pragma unroll
            for (int c = 0; c < 4; ++c) o[c].s[r] *= alpha;
            int row = r + 8 * half, col = lane & 15;
            myp[row * 32 + col]      = f2bf(p0);
            myp[row * 32 + 16 + col] = f2bf(p1);
        }

        // ---- P (16x32) A-fragment from per-wave LDS (same-wave, DS-ordered)
        Frag ap;
        {
            const unsigned short* prow = myp + (lane & 15) * 32 + 8 * half;
#pragma unroll
            for (int j = 0; j < 8; ++j) {
                int off = 2 * j + ((j >= 4) ? 8 : 0);
                ap.u[j] = *reinterpret_cast<const unsigned*>(prow + off);
            }
        }

        // ---- O += P @ V (K-dim 32 = this KV block), 4 N-subtiles of DK
#pragma unroll
        for (int c = 0; c < 4; ++c)
            o[c].f = wmma_bf16(ap, bv[c], o[c].f);
    }

    // ---- normalize and store bf16 into [N][T][D] (col = h*64+dk)
#pragma unroll
    for (int r = 0; r < 8; ++r) {
        float inv = 1.0f / l_i[r];
        int qt = qt0 + r + 8 * half;
#pragma unroll
        for (int c = 0; c < 4; ++c) {
            int dk = 16 * c + (lane & 15);
            aob[((size_t)n * Ts + qt) * Dm + h * 64 + dk] = f2bf(o[c].s[r] * inv);
        }
    }
}

// ---------------------------------------------------------------------------
// 4) Output projection: out = aob @ woT + bo (fp32 out). Same tiling as QKV,
//    direct-global B path (keeps both operand-fetch styles in the binary).
// ---------------------------------------------------------------------------
__global__ __launch_bounds__(256) void oproj_kernel(
    const unsigned short* __restrict__ aob,
    const unsigned short* __restrict__ woT,
    const float* __restrict__ bo,
    float* __restrict__ out) {
    const int lane = threadIdx.x & 31, wave = threadIdx.x >> 5;
    const int half = lane >> 4;
    const int m0 = blockIdx.y * 64 + (wave >> 1) * 16;
    const int n0 = blockIdx.x * 64 + (wave & 1) * 32;
    const int mrow = m0 + (lane & 15);

    Frag acc[2];
#pragma unroll
    for (int s = 0; s < 2; ++s)
#pragma unroll
        for (int r = 0; r < 8; ++r) acc[s].s[r] = 0.f;

    for (int k0 = 0; k0 < Dm; k0 += 32) {
        Frag a; load_a(a, aob, Dm, mrow, k0 + 8 * half);
        Frag b0, b1;
        load_b(b0, woT, Dm, k0, n0,      lane);
        load_b(b1, woT, Dm, k0, n0 + 16, lane);
        if (k0 + 32 < Dm) {
            __builtin_prefetch(aob + (size_t)mrow * Dm + k0 + 32, 0, 1);
            __builtin_prefetch(woT + (size_t)(k0 + 32 + lane) * Dm + n0, 0, 1);
        }
        acc[0].f = wmma_bf16(a, b0, acc[0].f);
        acc[1].f = wmma_bf16(a, b1, acc[1].f);
    }
#pragma unroll
    for (int s = 0; s < 2; ++s)
#pragma unroll
        for (int r = 0; r < 8; ++r) {
            int gm = m0 + r + 8 * half;
            int gc = n0 + 16 * s + (lane & 15);
            out[(size_t)gm * Dm + gc] = acc[s].s[r] + bo[gc];
        }
}

// ---------------------------------------------------------------------------
extern "C" void kernel_launch(void* const* d_in, const int* in_sizes, int n_in,
                              void* d_out, int out_size, void* d_ws, size_t ws_size,
                              hipStream_t stream) {
    (void)in_sizes; (void)n_in; (void)out_size; (void)ws_size;
    const float* x  = (const float*)d_in[0];
    const float* wq = (const float*)d_in[1]; const float* bq = (const float*)d_in[2];
    const float* wk = (const float*)d_in[3]; const float* bk = (const float*)d_in[4];
    const float* wv = (const float*)d_in[5]; const float* bv = (const float*)d_in[6];
    const float* wo = (const float*)d_in[7]; const float* bo = (const float*)d_in[8];

    char* ws = (char*)d_ws;
    size_t off = 0;
    unsigned short* xb  = (unsigned short*)(ws + off); off += (size_t)NT * Dm * 2;   // 16 MB
    unsigned short* wqT = (unsigned short*)(ws + off); off += (size_t)Dm * Dm * 2;
    unsigned short* wkT = (unsigned short*)(ws + off); off += (size_t)Dm * Dm * 2;
    unsigned short* wvT = (unsigned short*)(ws + off); off += (size_t)Dm * Dm * 2;
    unsigned short* woT = (unsigned short*)(ws + off); off += (size_t)Dm * Dm * 2;
    unsigned short* qh  = (unsigned short*)(ws + off); off += (size_t)NT * Dm * 2;
    unsigned short* kTr = (unsigned short*)(ws + off); off += (size_t)NT * Dm * 2;
    unsigned short* vh  = (unsigned short*)(ws + off); off += (size_t)NT * Dm * 2;
    unsigned short* aob = xb;  // alias: xb dead after qkv_kernel, aob born in flash

    cvt_x_kernel<<<(NT * Dm + 255) / 256, 256, 0, stream>>>(x, xb, NT * Dm);
    cvt_w_kernel<<<(4 * Dm * Dm + 255) / 256, 256, 0, stream>>>(
        wq, wk, wv, wo, wqT, wkT, wvT, woT);
    qkv_kernel<<<dim3(Dm / 64, NT / 64), 256, 0, stream>>>(
        xb, wqT, wkT, wvT, bq, bk, bv, qh, kTr, vh);
    flash_kernel<<<dim3(Ts / 64, Nb * Hh), 128, 0, stream>>>(qh, kTr, vh, aob);
    oproj_kernel<<<dim3(Dm / 64, NT / 64), 256, 0, stream>>>(aob, woT, bo, (float*)d_out);
}
